// GraphDiTPolicy_54039278519008
// MI455X (gfx1250) — matrix operand order repaired and checked by reading, not compiled, not run on hardware
//
#include <hip/hip_runtime.h>

// MI455X / gfx1250 fused GraphDiT attention.
// B=32768, N=2, T=4, D=256, H=8, E=2, HD=32, S=8.
// One wave handles 2 batches (16 stacked rows) so every GEMM maps onto
// v_wmma_f32_16x16x32_f16. d_ws holds Wq/Wk/Wv/Wo pre-converted to f16 in
// WMMA B-fragment order (512 KB).

typedef _Float16 half_t;
typedef __attribute__((ext_vector_type(16))) _Float16 v16h;
typedef __attribute__((ext_vector_type(8)))  float    v8f;

union FragH { v16h v; half_t h[16]; };

#define WMMA(A, B, C) \
  __builtin_amdgcn_wmma_f32_16x16x32_f16(false, (A), false, (B), (short)0, (C), false, false)

__device__ __host__ __forceinline__ size_t woff(int p, int nt, int kb, int lane) {
  // [proj][ntile 16][kstep 8][lane 32][16 halfs]
  return ((((size_t)p * 16 + nt) * 8 + kb) * 32 + lane) * 16;
}

// ---- Kernel 0: convert f32 weights -> f16 B-fragment layout in workspace ----
// B-fragment (32x16 f16): lane n=lane%16 holds column n; halfs j=0..15 hold
// K = 32*kb + 8*(lane/16) + (j%8) + 16*(j/8)  (table K order per lane).
__global__ __launch_bounds__(32) void prep_weights_kernel(
    const float* __restrict__ Wq, const float* __restrict__ Wk,
    const float* __restrict__ Wv, const float* __restrict__ Wo,
    half_t* __restrict__ wsw) {
  const int bid  = blockIdx.x;            // proj*128 + nt*8 + kb
  const int lane = threadIdx.x;
  const int proj = bid >> 7, nt = (bid >> 3) & 15, kb = bid & 7;
  const float* W = (proj == 0) ? Wq : (proj == 1) ? Wk : (proj == 2) ? Wv : Wo;
  const int n  = nt * 16 + (lane & 15);
  const int k0 = kb * 32 + ((lane >> 4) << 3);
  FragH f;
#pragma unroll
  for (int j = 0; j < 16; ++j) f.h[j] = (half_t)W[(size_t)(k0 + (j & 7) + ((j >> 3) << 4)) * 256 + n];
  *(v16h*)(wsw + woff(proj, nt, kb, lane)) = f.v;
}

// ---- Kernel 1: fused QKV + edge modulation + attention + pool + out proj ----
__global__ __launch_bounds__(256) void fused_attn_kernel(
    const float* __restrict__ nf,     const float* __restrict__ edge,
    const float* __restrict__ bq,     const float* __restrict__ bk,
    const float* __restrict__ bv,     const float* __restrict__ bo,
    const float* __restrict__ Wbias,  const float* __restrict__ bbias,
    const float* __restrict__ Wgate,  const float* __restrict__ bgate,
    const float* __restrict__ Wscale, const float* __restrict__ bscale,
    const float* __restrict__ Wshift, const float* __restrict__ bshift,
    const float* __restrict__ Whs,    const float* __restrict__ bhs,
    const half_t* __restrict__ wsw,   float* __restrict__ outp) {

  __shared__ half_t qbuf[8][16][32];   // per-wave Q head-slice (f16)
  __shared__ half_t kbuf[8][16][32];   // per-wave K head-slice (f16)
  __shared__ half_t vtb [8][32][16];   // per-wave V^T head-slice (f16)
  __shared__ float  scb [8][16][8];    // per-wave scores (valid 8 keys)
  __shared__ half_t atb [8][16][8];    // per-wave softmaxed attn
  __shared__ half_t pooled[32][256];   // block-wide pooled rows (f16)

  const int tid  = threadIdx.x;
  const int w    = tid >> 5;
  const int lane = tid & 31;
  const int ml   = lane & 15;          // A-frag row / B-frag & D-tile column
  const int kh   = lane >> 4;          // K-half selector (A/B frags)
  const int hf   = kh;                 // batch half for D tiles (rows 0-7 / 8-15)
  const float SCALE = 0.17677669529663687f;  // 1/sqrt(32)

  const int pair = blockIdx.x * 8 + w;       // batches 2*pair, 2*pair+1

  // per-lane edge features for the batch half this lane's D-tile rows belong to
  const float* ef = edge + (size_t)pair * 4;
  const float ea  = ef[hf ? 2 : 0];
  const float eb_ = ef[hf ? 3 : 1];

  // ---- A-fragments of x (16 stacked rows x 256), f32 -> f16, reused for Q/K/V
  // A layout: lane ml holds row ml; half j -> K = 32*kb + (j%8) + 16*(j/8) + 8*kh
  v16h afrag[8];
  {
    const float* xr = nf + ((size_t)pair * 16 + ml) * 256;
#pragma unroll
    for (int kb = 0; kb < 8; ++kb) {
      const int c0 = kb * 32 + 8 * kh;
      FragH f;
#pragma unroll
      for (int j = 0; j < 8; ++j) f.h[j]     = (half_t)xr[c0 + j];
#pragma unroll
      for (int j = 0; j < 8; ++j) f.h[8 + j] = (half_t)xr[c0 + 16 + j];
      afrag[kb] = f.v;
    }
  }

  const v8f vzero = {0.f, 0.f, 0.f, 0.f, 0.f, 0.f, 0.f, 0.f};

  for (int h = 0; h < 8; ++h) {
    // ---------- Q, K, V(modulated) tiles for this head (cols 32h..32h+31) ----
#pragma unroll
    for (int u = 0; u < 2; ++u) {
      const int nt = 2 * h + u;
      const int c  = nt * 16 + ml;

      v8f qa = vzero, ka = vzero, va = vzero;
#pragma unroll
      for (int kb = 0; kb < 8; ++kb) {
        const v16h wq = *(const v16h*)(wsw + woff(0, nt, kb, lane));
        qa = WMMA(afrag[kb], wq, qa);
        const v16h wk = *(const v16h*)(wsw + woff(1, nt, kb, lane));
        ka = WMMA(afrag[kb], wk, ka);
        const v16h wv = *(const v16h*)(wsw + woff(2, nt, kb, lane));
        va = WMMA(afrag[kb], wv, va);
      }
      const float bqv = bq[c], bkv = bk[c], bvv = bv[c];
      // edge-conditioned modulation: Vmod = V*am + cm, am=(1+tanh)*gate, cm=shift*gate
      const float g  = 1.f / (1.f + __expf(-(ea * Wgate[c] + eb_ * Wgate[256 + c] + bgate[c])));
      const float sz = tanhf(ea * Wscale[c] + eb_ * Wscale[256 + c] + bscale[c]);
      const float hz =       ea * Wshift[c] + eb_ * Wshift[256 + c] + bshift[c];
      const float am = (1.f + sz) * g, cm = hz * g;
#pragma unroll
      for (int r = 0; r < 8; ++r) {
        qbuf[w][r + 8 * hf][u * 16 + ml] = (half_t)(qa[r] + bqv);
        kbuf[w][r + 8 * hf][u * 16 + ml] = (half_t)(ka[r] + bkv);
        vtb [w][u * 16 + ml][r + 8 * hf] = (half_t)((va[r] + bvv) * am + cm);
      }
    }
    __syncthreads();

    // ---------- scores = (Q K^T) * SCALE * (1+head_scale) + sparse bias ------
    FragH qf, kf;
#pragma unroll
    for (int j = 0; j < 8; ++j) {
      qf.h[j]     = qbuf[w][ml][8 * kh + j];
      qf.h[8 + j] = qbuf[w][ml][8 * kh + 16 + j];
    }
#pragma unroll
    for (int j = 0; j < 16; ++j) kf.h[j] = kbuf[w][ml][8 * kh + (j & 7) + ((j >> 3) << 4)];
    v8f sa = vzero;
    sa = WMMA(qf.v, kf.v, sa);   // diagonal 8x8 blocks valid per batch

    const float hsb = tanhf(ea * Whs[h] + eb_ * Whs[8 + h] + bhs[h]);
    const float fac = SCALE * (1.f + hsb);
    const float ebb = ea * Wbias[h] + eb_ * Wbias[8 + h] + bbias[h];

    const bool valid = (lane < 8) || (lane >= 24);   // in-batch key columns only
    if (valid) {
      float srow[8];
#pragma unroll
      for (int r = 0; r < 8; ++r) srow[r] = sa[r] * fac;
      srow[(ml & 7) ^ 4] += ebb;                     // bias at s_q = s_k ^ 4
#pragma unroll
      for (int r = 0; r < 8; ++r) scb[w][r + 8 * hf][ml & 7] = srow[r];
    }
    __syncthreads();

    if (lane < 16) {                                 // one row per lane
      float vr[8], mx = -3.0e38f;
#pragma unroll
      for (int c2 = 0; c2 < 8; ++c2) { vr[c2] = scb[w][lane][c2]; mx = fmaxf(mx, vr[c2]); }
      float s = 0.f;
#pragma unroll
      for (int c2 = 0; c2 < 8; ++c2) { vr[c2] = __expf(vr[c2] - mx); s += vr[c2]; }
      const float inv = 1.f / s;
#pragma unroll
      for (int c2 = 0; c2 < 8; ++c2) atb[w][lane][c2] = (half_t)(vr[c2] * inv);
    }
    __syncthreads();

    // ---------- out = attn_blockdiag @ V, then temporal mean-pool ------------
    FragH af;
    const bool use = (kh == 0) == (ml < 8);          // block-diagonal placement
#pragma unroll
    for (int j = 0; j < 8; ++j) {
      af.h[j]     = use ? atb[w][ml][j] : (half_t)0.f;
      af.h[8 + j] = (half_t)0.f;                     // K=16..31 zero padding
    }
#pragma unroll
    for (int u = 0; u < 2; ++u) {
      FragH vf;
#pragma unroll
      for (int j = 0; j < 16; ++j)
        vf.h[j] = (j < 8) ? vtb[w][u * 16 + ml][8 * kh + j] : (half_t)0.f;
      v8f oa = vzero;
      oa = WMMA(af.v, vf.v, oa);
      const int col = 32 * h + 16 * u + ml;
      const float p0 = 0.25f * (oa[0] + oa[1] + oa[2] + oa[3]);  // node 0 (t=0..3)
      const float p1 = 0.25f * (oa[4] + oa[5] + oa[6] + oa[7]);  // node 1
      pooled[w * 4 + hf * 2 + 0][col] = (half_t)p0;
      pooled[w * 4 + hf * 2 + 1][col] = (half_t)p1;
    }
  }
  __syncthreads();

  // ---------- final projection: out = pooled @ Wo + bo (block cooperative) ----
#pragma unroll
  for (int i = 0; i < 4; ++i) {
    const int jid = w + 8 * i;                 // 32 (mt,nt) jobs over 8 waves
    const int mt = jid >> 4, ntq = jid & 15;
    v8f acc = vzero;
#pragma unroll
    for (int kb = 0; kb < 8; ++kb) {
      FragH fa;
      const half_t* pr = &pooled[mt * 16 + ml][0];
      const int c0 = kb * 32 + 8 * kh;
#pragma unroll
      for (int j = 0; j < 8; ++j) {
        fa.h[j]     = pr[c0 + j];
        fa.h[8 + j] = pr[c0 + 16 + j];
      }
      const v16h fb = *(const v16h*)(wsw + woff(3, ntq, kb, lane));
      acc = WMMA(fa.v, fb, acc);
    }
    const int col = ntq * 16 + ml;
    const float bov = bo[col];
#pragma unroll
    for (int r = 0; r < 8; ++r) {
      const int prow = mt * 16 + r + 8 * kh;   // pooled row = wave*4 + batchHalf*2 + node
      const int wsrc = prow >> 2, rem = prow & 3;
      const size_t batch = (size_t)blockIdx.x * 16 + (size_t)(wsrc * 2 + (rem >> 1));
      outp[(batch * 2 + (size_t)(rem & 1)) * 256 + col] = acc[r] + bov;
    }
  }
}

extern "C" void kernel_launch(void* const* d_in, const int* in_sizes, int n_in,
                              void* d_out, int out_size, void* d_ws, size_t ws_size,
                              hipStream_t stream) {
  (void)in_sizes; (void)n_in; (void)out_size; (void)ws_size;
  const float* nf     = (const float*)d_in[0];
  const float* edge   = (const float*)d_in[1];
  const float* Wq     = (const float*)d_in[2];
  const float* bq     = (const float*)d_in[3];
  const float* Wk     = (const float*)d_in[4];
  const float* bk     = (const float*)d_in[5];
  const float* Wv     = (const float*)d_in[6];
  const float* bv     = (const float*)d_in[7];
  const float* Wo     = (const float*)d_in[8];
  const float* bo     = (const float*)d_in[9];
  const float* Wbias  = (const float*)d_in[10];
  const float* bbias  = (const float*)d_in[11];
  const float* Wgate  = (const float*)d_in[12];
  const float* bgate  = (const float*)d_in[13];
  const float* Wscale = (const float*)d_in[14];
  const float* bscale = (const float*)d_in[15];
  const float* Wshift = (const float*)d_in[16];
  const float* bshift = (const float*)d_in[17];
  const float* Whs    = (const float*)d_in[18];
  const float* bhs    = (const float*)d_in[19];

  half_t* wsw = (half_t*)d_ws;                 // 512 KB: f16 fragment weights
  float*  out = (float*)d_out;

  prep_weights_kernel<<<512, 32, 0, stream>>>(Wq, Wk, Wv, Wo, wsw);
  fused_attn_kernel<<<2048, 256, 0, stream>>>(
      nf, edge, bq, bk, bv, bo, Wbias, bbias, Wgate, bgate,
      Wscale, bscale, Wshift, bshift, Whs, bhs, wsw, out);
}
